// CausalSelfAttention_12678743458302
// MI455X (gfx1250) — matrix operand order repaired and checked by reading, not compiled, not run on hardware
//
#include <hip/hip_runtime.h>

#define B_  4
#define T_  2048
#define C_  1024
#define H_  16
#define D_  64
#define BT_ (B_*T_)            // 8192

typedef __attribute__((ext_vector_type(16))) __bf16       v16bf;
typedef __attribute__((ext_vector_type(8)))  float        v8f;
typedef __attribute__((ext_vector_type(4)))  unsigned int u32x4;

union V16 { v16bf v; struct { u32x4 lo, hi; } q; };

__device__ __forceinline__ V16 ldpair(const __bf16* p0, const __bf16* p1) {
  V16 r;
  r.q.lo = *(const u32x4*)p0;
  r.q.hi = *(const u32x4*)p1;
  return r;
}

__device__ __forceinline__ v8f wmma_bf16(v16bf a, v16bf b, v8f c) {
  // D = A(16x32 bf16) * B(32x16 bf16) + C(16x16 f32)
  return __builtin_amdgcn_wmma_f32_16x16x32_bf16(false, a, false, b, (short)0, c, false, false);
}

// CDNA5 async global->LDS copy (tracked by ASYNCcnt)
__device__ __forceinline__ void async_ld_b128(unsigned lds_off, const void* gaddr) {
  asm volatile("global_load_async_to_lds_b128 %0, %1, off"
               :: "v"(lds_off), "v"(gaddr) : "memory");
}
__device__ __forceinline__ void wait_async0() {
  asm volatile("s_wait_asynccnt 0" ::: "memory");
}

// reductions across the 16-lane half-wave that holds one row of a C/D tile
__device__ __forceinline__ float redmax16(float x) {
  x = fmaxf(x, __shfl_xor(x, 1, 32));
  x = fmaxf(x, __shfl_xor(x, 2, 32));
  x = fmaxf(x, __shfl_xor(x, 4, 32));
  x = fmaxf(x, __shfl_xor(x, 8, 32));
  return x;
}
__device__ __forceinline__ float redsum16(float x) {
  x += __shfl_xor(x, 1, 32);
  x += __shfl_xor(x, 2, 32);
  x += __shfl_xor(x, 4, 32);
  x += __shfl_xor(x, 8, 32);
  return x;
}

// ---------------- conversion kernels ----------------

__global__ __launch_bounds__(256) void cvt_bf16_kernel(const float* __restrict__ in,
                                                       __bf16* __restrict__ out, int n) {
  int i = blockIdx.x * 256 + threadIdx.x;
  if (i < n) out[i] = (__bf16)in[i];
}

// Wt[n*C + k] = (bf16) W[k*C + n]   (so B-operand rows are contiguous in K)
__global__ __launch_bounds__(256) void cvt_wT_kernel(const float* __restrict__ w,
                                                     __bf16* __restrict__ wt) {
  int i = blockIdx.x * 256 + threadIdx.x;   // i = n*C + k
  int n = i >> 10;
  int k = i & (C_ - 1);
  wt[i] = (__bf16)w[(size_t)k * C_ + n];
}

// ---------------- fused QKV projection GEMM ----------------
// Block (8 waves) computes 128(M) x 64(N).  B chunk (64x32 bf16, 4KB) is staged
// into double-buffered LDS with async global->LDS DMA shared by all 8 waves.
// k-loop manually unrolled 2x (stage A = buffer0, stage B = buffer1); all four
// B tiles are ds-loaded before the WMMA quartet so waits overlap LDS latency.
// which = blockIdx.y: 0->Q (B,H,T,D), 1->K (B,H,T,D), 2->V transposed (B,H,D,T)

__global__ __launch_bounds__(256) void qkv_gemm_kernel(
    const __bf16* __restrict__ xb,
    const __bf16* __restrict__ wqT, const __bf16* __restrict__ wkT, const __bf16* __restrict__ wvT,
    const float* __restrict__ bq, const float* __restrict__ bk, const float* __restrict__ bv,
    __bf16* __restrict__ q, __bf16* __restrict__ k, __bf16* __restrict__ vt)
{
  __shared__ __align__(16) __bf16 sB[2][64 * 32];   // 2 x 4KB double buffer

  const int which = blockIdx.y;
  const __bf16* wT  = (which == 0) ? wqT : (which == 1) ? wkT : wvT;
  const float* bias = (which == 0) ? bq  : (which == 1) ? bk  : bv;

  const int tid  = threadIdx.x;
  const int wave = tid >> 5, lane = tid & 31;
  const int lh = lane >> 4, ll = lane & 15;
  const int nt = blockIdx.x & (C_ / 64 - 1);      // 16 n-tiles
  const int mt = blockIdx.x / (C_ / 64);          // 64 m-tiles of 128
  const int m0 = mt * 128 + wave * 16;
  const int n0 = nt * 64;

  // async-copy mapping: one b128 per thread per chunk
  const int cp_n = tid >> 2;                      // 0..63  (n_local)
  const int cp_k = (tid & 3) * 8;                 // 0,8,16,24 (k offset)
  const __bf16* bsrc = wT + (size_t)(n0 + cp_n) * C_ + cp_k;
  const unsigned soff0 = (unsigned)(uintptr_t)&sB[0][cp_n * 32 + cp_k];
  const unsigned soff1 = soff0 + 64 * 32 * 2;

  const v8f z8 = {0.f,0.f,0.f,0.f,0.f,0.f,0.f,0.f};
  v8f acc[4];
#pragma unroll
  for (int s = 0; s < 4; ++s) acc[s] = z8;

  const __bf16* arow = xb + (size_t)(m0 + ll) * C_;
  const __bf16* b0 = &sB[0][0];
  const __bf16* b1 = &sB[1][0];

  // prologue: DMA chunk 0, prefetch A chunk 0 into registers
  async_ld_b128(soff0, bsrc);
  V16 aA = ldpair(arow + lh * 16, arow + lh * 16 + 8);
  V16 aB;

  const int NIT = C_ / 32;                        // 32 k-chunks (even)
  for (int it = 0; it < NIT; it += 2) {
    // ---- stage A: consume buffer 0 (chunk it); DMA chunk it+1 into buffer 1
    wait_async0();
    __syncthreads();
    async_ld_b128(soff1, bsrc + (it + 1) * 32);   // it+1 < NIT always (NIT even)
    aB = ldpair(arow + (it + 1) * 32 + lh * 16,
                arow + (it + 1) * 32 + lh * 16 + 8);
    {
      V16 bt[4];
#pragma unroll
      for (int s = 0; s < 4; ++s) {
        const __bf16* brow = b0 + (s * 16 + ll) * 32 + lh * 16;
        bt[s] = ldpair(brow, brow + 8);           // 8x ds_load_b128 batched
      }
#pragma unroll
      for (int s = 0; s < 4; ++s)
        acc[s] = wmma_bf16(aA.v, bt[s].v, acc[s]);
    }

    // ---- stage B: consume buffer 1 (chunk it+1); DMA chunk it+2 into buffer 0
    wait_async0();
    __syncthreads();
    if (it + 2 < NIT) {
      async_ld_b128(soff0, bsrc + (it + 2) * 32);
      aA = ldpair(arow + (it + 2) * 32 + lh * 16,
                  arow + (it + 2) * 32 + lh * 16 + 8);
    }
    {
      V16 bt[4];
#pragma unroll
      for (int s = 0; s < 4; ++s) {
        const __bf16* brow = b1 + (s * 16 + ll) * 32 + lh * 16;
        bt[s] = ldpair(brow, brow + 8);
      }
#pragma unroll
      for (int s = 0; s < 4; ++s)
        acc[s] = wmma_bf16(aB.v, bt[s].v, acc[s]);
    }
  }

#pragma unroll
  for (int s = 0; s < 4; ++s) {
    const int n = n0 + s * 16 + ll;
    const float bn_ = bias[n];
    const int h = n >> 6, d = n & 63;
#pragma unroll
    for (int r = 0; r < 8; ++r) {
      const int m  = m0 + r + 8 * lh;
      const int bb = m >> 11;          // m / T_
      const int t  = m & (T_ - 1);
      const float val = acc[s][r] + bn_;
      if (which == 2)
        vt[(((size_t)(bb * H_ + h)) * D_ + d) * T_ + t] = (__bf16)val;
      else if (which == 1)
        k[(((size_t)(bb * H_ + h)) * T_ + t) * D_ + d] = (__bf16)val;
      else
        q[(((size_t)(bb * H_ + h)) * T_ + t) * D_ + d] = (__bf16)val;
    }
  }
}

// ---------------- flash attention (one wave = one 16-query block) ----------------

__global__ __launch_bounds__(256) void flash_attn_kernel(
    const __bf16* __restrict__ q, const __bf16* __restrict__ k,
    const __bf16* __restrict__ vt, __bf16* __restrict__ yb)
{
  __shared__ __align__(16) __bf16 sP[8][16 * 32];    // per-wave P tile (16x32 bf16)
  const int wave = threadIdx.x >> 5;
  const int lane = threadIdx.x & 31;
  const int lh = lane >> 4, ll = lane & 15;
  const int gw = blockIdx.x * 8 + wave;              // B*H*(T/16) = 8192 blocks
  const int tb = gw & (T_ / 16 - 1);
  const int bh = gw / (T_ / 16);
  const int t0 = tb * 16;

  const __bf16* qb = q  + (size_t)bh * T_ * D_;
  const __bf16* kb = k  + (size_t)bh * T_ * D_;
  const __bf16* vb = vt + (size_t)bh * D_ * T_;

  // Q block as two A-operand tiles (d-chunks 0..31 / 32..63)
  const __bf16* qrow = qb + (size_t)(t0 + ll) * D_;
  V16 qa0 = ldpair(qrow + lh * 16,      qrow + lh * 16 + 8);
  V16 qa1 = ldpair(qrow + 32 + lh * 16, qrow + 32 + lh * 16 + 8);

  const v8f z8 = {0.f,0.f,0.f,0.f,0.f,0.f,0.f,0.f};
  float rowm[8], rowl[8];
  v8f o[4];
#pragma unroll
  for (int r = 0; r < 8; ++r) { rowm[r] = -__builtin_inff(); rowl[r] = 0.f; }
#pragma unroll
  for (int d = 0; d < 4; ++d) o[d] = z8;

  const float scale = 0.125f;   // 1/sqrt(64)

  // prefetch K tiles for the first s-block
  V16 kc[4];
  {
    const __bf16* kr0 = kb + (size_t)ll * D_ + lh * 16;
    const __bf16* kr1 = kb + (size_t)(16 + ll) * D_ + lh * 16;
    kc[0] = ldpair(kr0,      kr0 + 8);
    kc[1] = ldpair(kr1,      kr1 + 8);
    kc[2] = ldpair(kr0 + 32, kr0 + 40);
    kc[3] = ldpair(kr1 + 32, kr1 + 40);
  }

  const int send = t0 + 15;
  for (int s0 = 0; s0 <= send; s0 += 32) {
    // V tiles depend only on s0: issue early so they cover the softmax chain
    V16 vv[4];
#pragma unroll
    for (int d = 0; d < 4; ++d) {
      const __bf16* vr = vb + (size_t)(d * 16 + ll) * T_ + s0 + lh * 16;
      vv[d] = ldpair(vr, vr + 8);
    }

    // ---- scores S = Q * K^T for keys [s0, s0+32): two 16x16 C tiles
    v8f c0 = z8, c1 = z8;
    c0 = wmma_bf16(qa0.v, kc[0].v, c0);
    c1 = wmma_bf16(qa0.v, kc[1].v, c1);
    c0 = wmma_bf16(qa1.v, kc[2].v, c0);
    c1 = wmma_bf16(qa1.v, kc[3].v, c1);

    // prefetch next s-block's K tiles while softmax runs
    if (s0 + 32 <= send) {
      const __bf16* kr0 = kb + (size_t)(s0 + 32 + ll) * D_ + lh * 16;
      const __bf16* kr1 = kb + (size_t)(s0 + 48 + ll) * D_ + lh * 16;
      kc[0] = ldpair(kr0,      kr0 + 8);
      kc[1] = ldpair(kr1,      kr1 + 8);
      kc[2] = ldpair(kr0 + 32, kr0 + 40);
      kc[3] = ldpair(kr1 + 32, kr1 + 40);
    }

    // ---- online softmax in C layout (row = (r, half), col = lane&15)
    const bool need_mask = (s0 + 31 > t0);
#pragma unroll
    for (int r = 0; r < 8; ++r) {
      const int trow = t0 + r + 8 * lh;
      float x0 = c0[r] * scale;
      float x1 = c1[r] * scale;
      if (need_mask) {
        if (s0 + ll      > trow) x0 = -__builtin_inff();
        if (s0 + 16 + ll > trow) x1 = -__builtin_inff();
      }
      float bm    = redmax16(fmaxf(x0, x1));
      float nm    = fmaxf(rowm[r], bm);
      float alpha = __expf(rowm[r] - nm);
      rowm[r] = nm;
      float p0 = __expf(x0 - nm);
      float p1 = __expf(x1 - nm);
      rowl[r] = rowl[r] * alpha + redsum16(p0 + p1);
#pragma unroll
      for (int d = 0; d < 4; ++d) o[d][r] *= alpha;
      // stage P into LDS (C layout -> memory row-major 16x32)
      sP[wave][(r + 8 * lh) * 32 + ll]      = (__bf16)p0;
      sP[wave][(r + 8 * lh) * 32 + 16 + ll] = (__bf16)p1;
    }

    asm volatile("s_wait_dscnt 0" ::: "memory");   // LDS RAW fence (CDNA5 split counter)

    // re-load P in A-operand layout
    const __bf16* pr = &sP[wave][ll * 32 + lh * 16];
    V16 pa = ldpair(pr, pr + 8);

    // ---- O += P * V  (V stored transposed: B operand rows contiguous in s)
#pragma unroll
    for (int d = 0; d < 4; ++d)
      o[d] = wmma_bf16(pa.v, vv[d].v, o[d]);
  }

  // ---- epilogue: y[b, t, h*64+d] = O / l  (bf16 for the output projection)
  const int bb = bh / H_;
  const int h  = bh & (H_ - 1);
#pragma unroll
  for (int d = 0; d < 4; ++d) {
#pragma unroll
    for (int r = 0; r < 8; ++r) {
      const int trow = t0 + r + 8 * lh;
      const float val = o[d][r] / rowl[r];
      yb[((size_t)(bb * T_ + trow)) * C_ + h * D_ + d * 16 + ll] = (__bf16)val;
    }
  }
}

// ---------------- output projection GEMM (fp32 out) ----------------
// Same block-cooperative async-LDS structure as the QKV GEMM.

__global__ __launch_bounds__(256) void proj_gemm_kernel(
    const __bf16* __restrict__ yb, const __bf16* __restrict__ wpT,
    const float* __restrict__ bp, float* __restrict__ out)
{
  __shared__ __align__(16) __bf16 sB[2][64 * 32];

  const int tid  = threadIdx.x;
  const int wave = tid >> 5, lane = tid & 31;
  const int lh = lane >> 4, ll = lane & 15;
  const int nt = blockIdx.x & (C_ / 64 - 1);
  const int mt = blockIdx.x / (C_ / 64);
  const int m0 = mt * 128 + wave * 16;
  const int n0 = nt * 64;

  const int cp_n = tid >> 2;
  const int cp_k = (tid & 3) * 8;
  const __bf16* bsrc = wpT + (size_t)(n0 + cp_n) * C_ + cp_k;
  const unsigned soff0 = (unsigned)(uintptr_t)&sB[0][cp_n * 32 + cp_k];
  const unsigned soff1 = soff0 + 64 * 32 * 2;

  const v8f z8 = {0.f,0.f,0.f,0.f,0.f,0.f,0.f,0.f};
  v8f acc[4];
#pragma unroll
  for (int s = 0; s < 4; ++s) acc[s] = z8;

  const __bf16* arow = yb + (size_t)(m0 + ll) * C_;
  const __bf16* b0 = &sB[0][0];
  const __bf16* b1 = &sB[1][0];

  async_ld_b128(soff0, bsrc);
  V16 aA = ldpair(arow + lh * 16, arow + lh * 16 + 8);
  V16 aB;

  const int NIT = C_ / 32;
  for (int it = 0; it < NIT; it += 2) {
    wait_async0();
    __syncthreads();
    async_ld_b128(soff1, bsrc + (it + 1) * 32);
    aB = ldpair(arow + (it + 1) * 32 + lh * 16,
                arow + (it + 1) * 32 + lh * 16 + 8);
    {
      V16 bt[4];
#pragma unroll
      for (int s = 0; s < 4; ++s) {
        const __bf16* brow = b0 + (s * 16 + ll) * 32 + lh * 16;
        bt[s] = ldpair(brow, brow + 8);
      }
#pragma unroll
      for (int s = 0; s < 4; ++s)
        acc[s] = wmma_bf16(aA.v, bt[s].v, acc[s]);
    }

    wait_async0();
    __syncthreads();
    if (it + 2 < NIT) {
      async_ld_b128(soff0, bsrc + (it + 2) * 32);
      aA = ldpair(arow + (it + 2) * 32 + lh * 16,
                  arow + (it + 2) * 32 + lh * 16 + 8);
    }
    {
      V16 bt[4];
#pragma unroll
      for (int s = 0; s < 4; ++s) {
        const __bf16* brow = b1 + (s * 16 + ll) * 32 + lh * 16;
        bt[s] = ldpair(brow, brow + 8);
      }
#pragma unroll
      for (int s = 0; s < 4; ++s)
        acc[s] = wmma_bf16(aB.v, bt[s].v, acc[s]);
    }
  }

#pragma unroll
  for (int s = 0; s < 4; ++s) {
    const int n = n0 + s * 16 + ll;
    const float bn_ = bp[n];
#pragma unroll
    for (int r = 0; r < 8; ++r) {
      const int m = m0 + r + 8 * lh;
      out[(size_t)m * C_ + n] = acc[s][r] + bn_;
    }
  }
}

// ---------------- launcher ----------------

extern "C" void kernel_launch(void* const* d_in, const int* in_sizes, int n_in,
                              void* d_out, int out_size, void* d_ws, size_t ws_size,
                              hipStream_t stream)
{
  const float* x  = (const float*)d_in[0];
  const float* Wq = (const float*)d_in[1];
  const float* bq = (const float*)d_in[2];
  const float* Wk = (const float*)d_in[3];
  const float* bk = (const float*)d_in[4];
  const float* Wv = (const float*)d_in[5];
  const float* bv = (const float*)d_in[6];
  const float* Wp = (const float*)d_in[7];
  const float* bp = (const float*)d_in[8];
  float* out = (float*)d_out;

  char* ws = (char*)d_ws;
  size_t off = 0;
  auto alloc = [&](size_t bytes) {
    char* p = ws + off;
    off += (bytes + 255) & ~(size_t)255;
    return p;
  };
  __bf16* xb  = (__bf16*)alloc((size_t)BT_ * C_ * 2);   // 16 MB
  __bf16* wqT = (__bf16*)alloc((size_t)C_ * C_ * 2);    //  2 MB
  __bf16* wkT = (__bf16*)alloc((size_t)C_ * C_ * 2);
  __bf16* wvT = (__bf16*)alloc((size_t)C_ * C_ * 2);
  __bf16* wpT = (__bf16*)alloc((size_t)C_ * C_ * 2);
  __bf16* qb  = (__bf16*)alloc((size_t)BT_ * C_ * 2);   // (B,H,T,D)
  __bf16* kbf = (__bf16*)alloc((size_t)BT_ * C_ * 2);   // (B,H,T,D)
  __bf16* vtb = (__bf16*)alloc((size_t)BT_ * C_ * 2);   // (B,H,D,T)
  __bf16* ybf = (__bf16*)alloc((size_t)BT_ * C_ * 2);   // (B,T,C)

  const int nx = BT_ * C_;                              // 8388608
  cvt_bf16_kernel<<<nx / 256, 256, 0, stream>>>(x, xb, nx);
  const int nw = C_ * C_;                               // 1048576
  cvt_wT_kernel<<<nw / 256, 256, 0, stream>>>(Wq, wqT);
  cvt_wT_kernel<<<nw / 256, 256, 0, stream>>>(Wk, wkT);
  cvt_wT_kernel<<<nw / 256, 256, 0, stream>>>(Wv, wvT);
  cvt_wT_kernel<<<nw / 256, 256, 0, stream>>>(Wp, wpT);

  dim3 gq((BT_ / 128) * (C_ / 64), 3);                  // 1024 x 3
  qkv_gemm_kernel<<<gq, 256, 0, stream>>>(xb, wqT, wkT, wvT, bq, bk, bv, qb, kbf, vtb);

  flash_attn_kernel<<<(B_ * H_ * (T_ / 16)) / 8, 256, 0, stream>>>(qb, kbf, vtb, ybf);

  proj_gemm_kernel<<<(BT_ / 128) * (C_ / 64), 256, 0, stream>>>(ybf, wpT, bp, out);
}
